// InteractionMap_66348654788735
// MI455X (gfx1250) — matrix operand order repaired and checked by reading, not compiled, not run on hardware
//
#include <hip/hip_runtime.h>
#include <hip/hip_bf16.h>

typedef __attribute__((ext_vector_type(16))) _Float16 v16h;
typedef __attribute__((ext_vector_type(8)))  _Float16 v8h;
typedef __attribute__((ext_vector_type(8)))  float    v8f;
// 16-byte-aligned view of a 32-byte vector: max VMEM/DS access is b128, so 16B
// alignment is sufficient and lets us read 80B-strided LDS rows.
typedef v16h uv16h __attribute__((aligned(16)));

constexpr int B_  = 2;
constexpr int S_  = 2048;
constexpr int H_  = 1024;
constexpr int NH_ = 16;
constexpr int DH_ = 64;
constexpr int E_  = 128;
constexpr int R_  = 64;

// ---------------------------------------------------------------------------
// utility kernels
// ---------------------------------------------------------------------------
__global__ void cast_f2h(const float* __restrict__ src, _Float16* __restrict__ dst, int n) {
    int i = blockIdx.x * 256 + threadIdx.x;
    if (i < n) dst[i] = (_Float16)src[i];
}

// f32 -> f16 with no k-permutation inside every
// aligned 32-element block of the innermost dim (8-blocks keep their place):
// a lane's 16 A-frag halves become one contiguous 32B chunk.
__global__ void cast_permute_a(const float* __restrict__ src, _Float16* __restrict__ dst, int n) {
    int i = blockIdx.x * 256 + threadIdx.x;
    if (i >= n) return;
    int d32 = i & 31;
    int g   = d32 >> 3;                     // 8-block index 0..3
    int gp  = g;
    int j   = (i & ~31) | (gp * 8 + (d32 & 7));
    dst[j] = (_Float16)src[i];
}

__global__ void gather_cast(const float* __restrict__ hidden, const int* __restrict__ pos,
                            _Float16* __restrict__ dst, int cnt) {
    int i = blockIdx.x * 256 + threadIdx.x;
    if (i >= cnt * H_) return;
    int r = i / H_, c = i - r * H_;
    dst[i] = (_Float16)hidden[(size_t)pos[r] * H_ + c];   // batch 0
}

// W (K,N) f32 row-major -> Wt (N,K) f16 : B-fragments become contiguous loads
__global__ void transpose_cast_w(const float* __restrict__ W, _Float16* __restrict__ Wt,
                                 int K, int N) {
    int i = blockIdx.x * 256 + threadIdx.x;
    if (i >= K * N) return;
    int k = i / N, n = i - k * N;
    Wt[(size_t)n * K + k] = (_Float16)W[i];
}

// V (b,s,h,d) f32  ->  Vt (b,h,d,s) f16
__global__ void transpose_cast_v(const float* __restrict__ V, _Float16* __restrict__ Vt, int total) {
    int i = blockIdx.x * 256 + threadIdx.x;
    if (i >= total) return;
    int b   = i / (S_ * H_);
    int rem = i - b * (S_ * H_);
    int s   = rem / H_;
    int c   = rem - s * H_;
    int h   = c / DH_;
    int d   = c - h * DH_;
    Vt[(((size_t)b * NH_ + h) * DH_ + d) * S_ + s] = (_Float16)V[i];
}

// ---------------------------------------------------------------------------
// WMMA tiled GEMM:  C[M,N] (f32, +bias) = A[M,K] (f16 row-major) @ Bt[N,K]^T (f16)
// block = 256 threads = 8 waves, block tile 128(M) x 64(N), wave tile 32x32, K step 32
// A staged in LDS in plain k order so frags are contiguous b128 reads;
// B frags read straight from the L2-resident transposed weights (contiguous 32B).
// ---------------------------------------------------------------------------
__global__ __launch_bounds__(256) void gemm_f16f32(const _Float16* __restrict__ A,
                                                   const _Float16* __restrict__ Bt,
                                                   const float* __restrict__ bias,
                                                   float* __restrict__ C,
                                                   int M, int N, int K) {
    __shared__ _Float16 As[128][40];   // 32 K (plain order) + 8 pad

    const int tid  = threadIdx.x;
    const int lane = tid & 31, wid = tid >> 5;
    const int waveM = wid & 3, waveN = wid >> 2;       // 4 x 2 wave grid
    const int l16 = lane & 15, half = lane >> 4;
    const int bm = blockIdx.x * 128, bn = blockIdx.y * 64;

    v8f acc[2][2];
#pragma unroll
    for (int mt = 0; mt < 2; ++mt)
#pragma unroll
        for (int nt = 0; nt < 2; ++nt)
#pragma unroll
            for (int r = 0; r < 8; ++r) acc[mt][nt][r] = 0.f;

    const _Float16* bbase[2];
#pragma unroll
    for (int nt = 0; nt < 2; ++nt)
        bbase[nt] = Bt + (size_t)(bn + waveN * 32 + nt * 16 + l16) * K + 16 * half;

    for (int k0 = 0; k0 < K; k0 += 32) {
        // ---- stage A block tile 128x32 (16 halves / thread), plain k order per row
        {
            int row = tid >> 1, seg = tid & 1;
            int gr = bm + row;
            v8h a0, a1;
            if (gr < M) {
                const _Float16* src = A + (size_t)gr * K + k0 + seg * 16;
                a0 = *(const v8h*)(src);
                a1 = *(const v8h*)(src + 8);
                if (k0 + 32 < K) __builtin_prefetch(src + 32, 0, 0);
            } else {
#pragma unroll
                for (int i = 0; i < 8; ++i) { a0[i] = (_Float16)0.f; a1[i] = (_Float16)0.f; }
            }
            // no permutation: k seg*16+0..15 -> pos seg*16+0..15
            *(v8h*)&As[row][16 * seg]     = a0;
            *(v8h*)&As[row][16 * seg + 8] = a1;
        }
        __syncthreads();

        // ---- fragments: single contiguous vector reads, no packing moves
        v16h af[2], bf[2];
#pragma unroll
        for (int mt = 0; mt < 2; ++mt)
            af[mt] = *(const uv16h*)&As[waveM * 32 + mt * 16 + l16][16 * half];
#pragma unroll
        for (int nt = 0; nt < 2; ++nt)
            bf[nt] = *(const uv16h*)(bbase[nt] + k0);

#pragma unroll
        for (int mt = 0; mt < 2; ++mt)
#pragma unroll
            for (int nt = 0; nt < 2; ++nt)
                acc[mt][nt] = __builtin_amdgcn_wmma_f32_16x16x32_f16(
                    false, af[mt], false, bf[nt], (short)0, acc[mt][nt], false, false);
        __syncthreads();
    }

    // ---- epilogue (C layout: lane element r -> m = r + 8*half, n = lane%16)
#pragma unroll
    for (int mt = 0; mt < 2; ++mt)
#pragma unroll
        for (int nt = 0; nt < 2; ++nt) {
            int n = bn + waveN * 32 + nt * 16 + l16;
            float bv = bias ? bias[n] : 0.f;
#pragma unroll
            for (int r = 0; r < 8; ++r) {
                int m = bm + waveM * 32 + mt * 16 + r + 8 * half;
                if (m < M) C[(size_t)m * N + n] = acc[mt][nt][r] + bv;
            }
        }
}

// ---------------------------------------------------------------------------
// Flash attention, one wave per 16-query tile, 32 keys per step.
// Qp f16 (b,s,h,d) plain (no k-permutation); K f16 plain (b,s,h,d);
// V pre-transposed f16 (b,h,d,s). Output ctx f16 (b,s,h,d) plain.
// ---------------------------------------------------------------------------
__global__ __launch_bounds__(256) void flash_attn(const _Float16* __restrict__ Qp,
                                                  const _Float16* __restrict__ Kf,
                                                  const _Float16* __restrict__ Vt,
                                                  const float* __restrict__ mask,
                                                  _Float16* __restrict__ ctx) {
    __shared__ _Float16 Plds[8][16][40];   // per-wave 16x32 P tile (plain k order, +pad)

    const int tid  = threadIdx.x;
    const int lane = tid & 31, wid = tid >> 5;
    const int l16 = lane & 15, half = lane >> 4;

    int wgid = blockIdx.x * 8 + wid;
    const int QT = S_ / 16;                 // 128 query tiles per (b,h)
    int qt = wgid % QT;
    int bh = wgid / QT;
    int h  = bh % NH_;
    int b  = bh / NH_;

    // Q fragments: contiguous 32B reads of the plain layout
    const _Float16* qbase = Qp + ((size_t)(b * S_ + qt * 16 + l16) * H_) + h * DH_;
    v16h qf[2];
#pragma unroll
    for (int f = 0; f < 2; ++f) qf[f] = *(const uv16h*)(qbase + 32 * f + 16 * half);

    v8f O[4];
#pragma unroll
    for (int t = 0; t < 4; ++t)
#pragma unroll
        for (int r = 0; r < 8; ++r) O[t][r] = 0.f;
    float Mrow[8], Lsum[8];
#pragma unroll
    for (int r = 0; r < 8; ++r) { Mrow[r] = -1e30f; Lsum[r] = 0.f; }

    const float scale = 0.125f;             // 1/sqrt(64)

    for (int kc = 0; kc < S_; kc += 32) {
        // ---- scores: two 16x16 tiles (keys kc..+15 / kc+16..+31)
        v8f sc[2];
#pragma unroll
        for (int t = 0; t < 2; ++t) {
            int key = kc + t * 16 + l16;
            const _Float16* kb = Kf + ((size_t)(b * S_ + key) * H_) + h * DH_;
            v8f s;
#pragma unroll
            for (int r = 0; r < 8; ++r) s[r] = 0.f;
#pragma unroll
            for (int f = 0; f < 2; ++f) {
                v16h bfr = *(const uv16h*)(kb + 32 * f + 16 * half);  // contiguous B-frag
                s = __builtin_amdgcn_wmma_f32_16x16x32_f16(
                    false, qf[f], false, bfr, (short)0, s, false, false);
            }
            float madd = (1.0f - mask[b * S_ + key]) * -10000.0f;
#pragma unroll
            for (int r = 0; r < 8; ++r) sc[t][r] = s[r] * scale + madd;
        }

        // ---- online softmax (row stats live in 16-lane groups)
        float nmax[8], alpha[8], p0[8], p1[8];
#pragma unroll
        for (int r = 0; r < 8; ++r) {
            float m2 = fmaxf(sc[0][r], sc[1][r]);
#pragma unroll
            for (int offm = 1; offm < 16; offm <<= 1) m2 = fmaxf(m2, __shfl_xor(m2, offm, 32));
            nmax[r] = fmaxf(Mrow[r], m2);
            alpha[r] = __expf(Mrow[r] - nmax[r]);
            p0[r] = __expf(sc[0][r] - nmax[r]);
            p1[r] = __expf(sc[1][r] - nmax[r]);
            float rs = p0[r] + p1[r];
#pragma unroll
            for (int offm = 1; offm < 16; offm <<= 1) rs += __shfl_xor(rs, offm, 32);
            Lsum[r] = Lsum[r] * alpha[r] + rs;
            Mrow[r] = nmax[r];
        }
#pragma unroll
        for (int t = 0; t < 4; ++t)
#pragma unroll
            for (int r = 0; r < 8; ++r) O[t][r] *= alpha[r];

        // ---- P (C-layout) -> LDS, stored in plain k order so the A-frag read is contiguous
        // col c goes to pos c
        int d0 = l16;                                // col = l16
        int d1 = 16 + l16;                           // col = 16 + l16
#pragma unroll
        for (int r = 0; r < 8; ++r) {
            int m = r + 8 * half;
            Plds[wid][m][d0] = (_Float16)p0[r];
            Plds[wid][m][d1] = (_Float16)p1[r];
        }
        asm volatile("s_wait_dscnt 0" ::: "memory");
        v16h pf = *(const uv16h*)&Plds[wid][l16][16 * half];

        // ---- O += P @ V  (V B-frags contiguous along s)
#pragma unroll
        for (int nt = 0; nt < 4; ++nt) {
            int d = nt * 16 + l16;
            const _Float16* vb = Vt + (((size_t)(b * NH_ + h) * DH_ + d) * S_) + kc + 16 * half;
            v16h vf = *(const uv16h*)(vb);
            O[nt] = __builtin_amdgcn_wmma_f32_16x16x32_f16(
                false, pf, false, vf, (short)0, O[nt], false, false);
        }
    }

    // ---- normalize + store ctx (f16, plain layout for the Wo GEMM)
#pragma unroll
    for (int nt = 0; nt < 4; ++nt) {
        int d = nt * 16 + l16;
#pragma unroll
        for (int r = 0; r < 8; ++r) {
            int m = r + 8 * half;
            ctx[((size_t)(b * S_ + qt * 16 + m)) * H_ + h * DH_ + d] = (_Float16)(O[nt][r] / Lsum[r]);
        }
    }
}

// ---------------------------------------------------------------------------
// residual + LayerNorm, one block per token row
// ---------------------------------------------------------------------------
__global__ __launch_bounds__(256) void ln_residual(const float* __restrict__ attn,
                                                   const float* __restrict__ hid,
                                                   const float* __restrict__ g,
                                                   const float* __restrict__ bb,
                                                   float* __restrict__ out) {
    __shared__ float red[256];
    int row = blockIdx.x, tid = threadIdx.x;
    const float* pa = attn + (size_t)row * H_;
    const float* ph = hid  + (size_t)row * H_;
    float s = 0.f, s2 = 0.f;
    for (int c = tid; c < H_; c += 256) {
        float x = pa[c] + ph[c];
        s += x; s2 += x * x;
    }
    red[tid] = s; __syncthreads();
    for (int st = 128; st > 0; st >>= 1) { if (tid < st) red[tid] += red[tid + st]; __syncthreads(); }
    float mean = red[0] / (float)H_;
    __syncthreads();
    red[tid] = s2; __syncthreads();
    for (int st = 128; st > 0; st >>= 1) { if (tid < st) red[tid] += red[tid + st]; __syncthreads(); }
    float var  = red[0] / (float)H_ - mean * mean;
    float rstd = rsqrtf(var + 1e-5f);
    for (int c = tid; c < H_; c += 256) {
        float x = pa[c] + ph[c];
        out[(size_t)row * H_ + c] = (x - mean) * rstd * g[c] + bb[c];
    }
}

// ---------------------------------------------------------------------------
// pair scores: out[i,j] = sum_c relu(a[i,c] + b[j,c] + b1[c]) * w2[c] + b2
// ---------------------------------------------------------------------------
__global__ __launch_bounds__(256) void pair_score(const float* __restrict__ Aa,
                                                  const float* __restrict__ Bb,
                                                  const float* __restrict__ b1,
                                                  const float* __restrict__ w2,
                                                  const float* __restrict__ b2,
                                                  float* __restrict__ out,
                                                  int NI, int NJ, int skipDiag) {
    int lane = threadIdx.x & 31, wid = threadIdx.x >> 5;
    int gid = blockIdx.x * 8 + wid;
    if (gid >= NI * NJ) return;
    int i = gid / NJ, j = gid - i * NJ;
    const float* pa = Aa + (size_t)i * H_;
    const float* pb = Bb + (size_t)j * H_;
    float s = 0.f;
    for (int c = lane; c < H_; c += 32) {
        float v = pa[c] + pb[c] + b1[c];
        s += fmaxf(v, 0.f) * w2[c];
    }
#pragma unroll
    for (int offm = 1; offm < 32; offm <<= 1) s += __shfl_xor(s, offm, 32);
    if (lane == 0) {
        if (skipDiag) {
            if (j != i) {
                int col = (j < i) ? j : (j - 1);
                out[(size_t)i * (NJ - 1) + col] = s + b2[0];
            }
        } else {
            out[(size_t)i * NJ + j] = s + b2[0];
        }
    }
}

// ---------------------------------------------------------------------------
extern "C" void kernel_launch(void* const* d_in, const int* in_sizes, int n_in,
                              void* d_out, int out_size, void* d_ws, size_t ws_size,
                              hipStream_t stream) {
    (void)in_sizes; (void)n_in; (void)out_size; (void)ws_size;
    const float* hidden = (const float*)d_in[0];
    const float* amask  = (const float*)d_in[1];
    const int*   epos   = (const int*)d_in[2];
    const int*   rpos   = (const int*)d_in[3];
    const float* Wq = (const float*)d_in[4];  const float* bq = (const float*)d_in[5];
    const float* Wk = (const float*)d_in[6];  const float* bk = (const float*)d_in[7];
    const float* Wv = (const float*)d_in[8];  const float* bv = (const float*)d_in[9];
    const float* Wo = (const float*)d_in[10]; const float* bo = (const float*)d_in[11];
    const float* ln_g = (const float*)d_in[12]; const float* ln_b = (const float*)d_in[13];
    const float* eeW1 = (const float*)d_in[14]; const float* eeb1 = (const float*)d_in[15];
    const float* eeW2 = (const float*)d_in[16]; const float* eeb2 = (const float*)d_in[17];
    const float* erW1 = (const float*)d_in[18]; const float* erb1 = (const float*)d_in[19];
    const float* erW2 = (const float*)d_in[20]; const float* erb2 = (const float*)d_in[21];

    const size_t NTOK = (size_t)B_ * S_;      // 4096 token rows
    const size_t NELT = NTOK * H_;            // 4,194,304 elements
    const size_t WELT = (size_t)H_ * H_;      // 1,048,576 weight elements

    char* ws = (char*)d_ws;
    size_t off = 0;
    auto alloc = [&](size_t bytes) -> void* {
        void* p = ws + off;
        off += (bytes + 255) & ~(size_t)255;
        return p;
    };
    _Float16* h16    = (_Float16*)alloc(NELT * 2);
    _Float16* wq16t  = (_Float16*)alloc(WELT * 2);   // all weights transposed (N,K) f16
    _Float16* wk16t  = (_Float16*)alloc(WELT * 2);
    _Float16* wv16t  = (_Float16*)alloc(WELT * 2);
    _Float16* wo16t  = (_Float16*)alloc(WELT * 2);
    _Float16* eeW1t  = (_Float16*)alloc(2 * WELT * 2);
    _Float16* erW1t  = (_Float16*)alloc(2 * WELT * 2);
    _Float16* eh16   = (_Float16*)alloc((size_t)E_ * H_ * 2);
    _Float16* rh16   = (_Float16*)alloc((size_t)R_ * H_ * 2);
    float*    Q32    = (float*)alloc(NELT * 4);
    float*    K32    = (float*)alloc(NELT * 4);
    float*    V32    = (float*)alloc(NELT * 4);
    _Float16* Qp16   = (_Float16*)alloc(NELT * 2);
    _Float16* K16    = (_Float16*)alloc(NELT * 2);
    _Float16* Vt16   = (_Float16*)alloc(NELT * 2);
    _Float16* ctx16  = (_Float16*)alloc(NELT * 2);
    float*    a_e    = (float*)alloc((size_t)E_ * H_ * 4);
    float*    b_e    = (float*)alloc((size_t)E_ * H_ * 4);
    float*    a_r    = (float*)alloc((size_t)E_ * H_ * 4);
    float*    b_r    = (float*)alloc((size_t)R_ * H_ * 4);
    float*    attn32 = Q32;   // Q32 free after Qp16 cast — reuse for output projection

    float* out_attn = (float*)d_out;
    float* out_ee   = out_attn + NELT;
    float* out_er   = out_ee + (size_t)E_ * (E_ - 1);

    auto cast = [&](const float* s, _Float16* d, int n) {
        cast_f2h<<<dim3((n + 255) / 256), dim3(256), 0, stream>>>(s, d, n);
    };
    auto wtrans = [&](const float* s, _Float16* d) {
        transpose_cast_w<<<dim3((int)((WELT + 255) / 256)), dim3(256), 0, stream>>>(s, d, H_, H_);
    };
    auto gemm = [&](const _Float16* A, const _Float16* Bt, const float* bias, float* C,
                    int M, int N, int K) {
        dim3 grid((M + 127) / 128, N / 64);
        gemm_f16f32<<<grid, dim3(256), 0, stream>>>(A, Bt, bias, C, M, N, K);
    };

    // 1) precision conversion + weight transposition
    cast(hidden, h16, (int)NELT);
    wtrans(Wq, wq16t); wtrans(Wk, wk16t); wtrans(Wv, wv16t); wtrans(Wo, wo16t);
    wtrans(eeW1, eeW1t);               wtrans(eeW1 + WELT, eeW1t + WELT);
    wtrans(erW1, erW1t);               wtrans(erW1 + WELT, erW1t + WELT);
    gather_cast<<<dim3((E_ * H_ + 255) / 256), dim3(256), 0, stream>>>(hidden, epos, eh16, E_);
    gather_cast<<<dim3((R_ * H_ + 255) / 256), dim3(256), 0, stream>>>(hidden, rpos, rh16, R_);

    // 2) QKV projections (WMMA GEMM)
    gemm(h16, wq16t, bq, Q32, (int)NTOK, H_, H_);
    gemm(h16, wk16t, bk, K32, (int)NTOK, H_, H_);
    gemm(h16, wv16t, bv, V32, (int)NTOK, H_, H_);

    // 3) attention-operand layouts
    cast_permute_a<<<dim3((int)((NELT + 255) / 256)), dim3(256), 0, stream>>>(Q32, Qp16, (int)NELT);
    cast(K32, K16, (int)NELT);
    transpose_cast_v<<<dim3((int)((NELT + 255) / 256)), dim3(256), 0, stream>>>(V32, Vt16, (int)NELT);

    // 4) flash attention (B*NH*S/16 waves, 8 waves/block)
    {
        int nwaves = B_ * NH_ * (S_ / 16);   // 4096
        flash_attn<<<dim3(nwaves / 8), dim3(256), 0, stream>>>(Qp16, K16, Vt16, amask, ctx16);
    }

    // 5) output projection + residual + LayerNorm
    gemm(ctx16, wo16t, bo, attn32, (int)NTOK, H_, H_);
    ln_residual<<<dim3((int)NTOK), dim3(256), 0, stream>>>(attn32, hidden, ln_g, ln_b, out_attn);

    // 6) interaction features (WMMA GEMMs; relu comes after the broadcast sum)
    gemm(eh16, eeW1t,        nullptr, a_e, E_, H_, H_);
    gemm(eh16, eeW1t + WELT, nullptr, b_e, E_, H_, H_);
    gemm(eh16, erW1t,        nullptr, a_r, E_, H_, H_);
    gemm(rh16, erW1t + WELT, nullptr, b_r, R_, H_, H_);

    // 7) pair-score epilogues
    pair_score<<<dim3((E_ * E_) / 8), dim3(256), 0, stream>>>(a_e, b_e, eeb1, eeW2, eeb2,
                                                              out_ee, E_, E_, 1);
    pair_score<<<dim3((E_ * R_) / 8), dim3(256), 0, stream>>>(a_r, b_r, erb1, erW2, erb2,
                                                              out_er, E_, R_, 0);
}
